// BlastRadiusGNN_30906584662489
// MI455X (gfx1250) — matrix operand order repaired and checked by reading, not compiled, not run on hardware
//
#include <hip/hip_runtime.h>
#include <string.h>

#define TPB 256

typedef __attribute__((ext_vector_type(16))) _Float16 v16h;
typedef __attribute__((ext_vector_type(8)))  _Float16 v8h;
typedef __attribute__((ext_vector_type(8)))  float    v8f;

// ---------- ordered-float encoding for atomic max on u32 ----------
__device__ __forceinline__ unsigned fenc(float f) {
    unsigned u = __float_as_uint(f);
    return (u & 0x80000000u) ? ~u : (u | 0x80000000u);
}
__device__ __forceinline__ float fdec(unsigned u) {
    return (u & 0x80000000u) ? __uint_as_float(u & 0x7fffffffu)
                             : __uint_as_float(~u);
}

// ---------- generic fills ----------
__global__ void k_fill_f32(float* p, float v, int n) {
    int i = blockIdx.x * blockDim.x + threadIdx.x;
    if (i < n) p[i] = v;
}
__global__ void k_fill_u32(unsigned* p, unsigned v, int n) {
    int i = blockIdx.x * blockDim.x + threadIdx.x;
    if (i < n) p[i] = v;
}

// ---------- self-loop attr: deg + segment-sum of edge_attr over dst ----------
__global__ void k_deg_attr(const int* dst, const float* ea, float* deg,
                           float* lattr, int E) {
    int e = blockIdx.x * blockDim.x + threadIdx.x;
    if (e >= E) return;
    int d = dst[e];
    atomicAdd(&deg[d], 1.0f);
    atomicAdd(&lattr[2 * d + 0], ea[2 * e + 0]);
    atomicAdd(&lattr[2 * d + 1], ea[2 * e + 1]);
}
__global__ void k_loop_div(float* lattr, const float* deg, int n) {
    int i = blockIdx.x * blockDim.x + threadIdx.x;
    if (i >= n) return;
    float dg = fmaxf(deg[i], 1.0f);
    lattr[2 * i + 0] /= dg;
    lattr[2 * i + 1] /= dg;
}

// ---------- plain node transform (tiny-K or tiny-N layers) ----------
__global__ void k_xform(const float* X, const float* W, float* Y,
                        int n, int IN, int OUT) {
    int i = blockIdx.x * blockDim.x + threadIdx.x;
    if (i >= n * OUT) return;
    int node = i / OUT, o = i % OUT;
    float s = 0.f;
    for (int k = 0; k < IN; ++k) s += X[node * IN + k] * W[k * OUT + o];
    Y[i] = s;
}

// ---------- attention coefficients per (node, head) ----------
template <int H, int C>
__global__ void k_alcoef(const float* __restrict__ T, const float* __restrict__ aS,
                         const float* __restrict__ aD, float* alS, float* alD, int n) {
    int i = blockIdx.x * blockDim.x + threadIdx.x;
    if (i >= n * H) return;
    int node = i / H, h = i % H;
    const float* t = T + node * H * C + h * C;
    float s = 0.f, d = 0.f;
#pragma unroll
    for (int c = 0; c < C; ++c) {
        s += t[c] * aS[h * C + c];
        d += t[c] * aD[h * C + c];
    }
    alS[i] = s;
    alD[i] = d;
}

// ---------- weV[k*H+h] = sum_c We[k, h*C+c] * aE[h, c]  (tiny) ----------
__global__ void k_wevec(const float* We, const float* aE, float* weV,
                        int H, int C) {
    int t = threadIdx.x;
    if (t >= 2 * H) return;
    int k = t / H, h = t % H;
    float s = 0.f;
    for (int c = 0; c < C; ++c) s += We[k * H * C + h * C + c] * aE[h * C + c];
    weV[k * H + h] = s;
}

// ---------- per-edge alpha (leaky-relu'd attention logit) ----------
template <int H>
__device__ __forceinline__ float gat_alpha(int e, int h, int E,
        const int* src, const int* dst, const float* ea, const float* lattr,
        const float* alS, const float* alD, const float* weV,
        int& s, int& d) {
    float a0, a1;
    if (e < E) {
        s = src[e]; d = dst[e];
        a0 = ea[2 * e + 0]; a1 = ea[2 * e + 1];
    } else {
        s = d = e - E;
        a0 = lattr[2 * s + 0]; a1 = lattr[2 * s + 1];
    }
    float al = alS[s * H + h] + alD[d * H + h] + a0 * weV[h] + a1 * weV[H + h];
    return al >= 0.f ? al : 0.2f * al;   // leaky_relu, slope 0.2
}

template <int H>
__global__ void k_edge_max(const int* __restrict__ src, const int* __restrict__ dst,
                           const float* __restrict__ ea, const float* __restrict__ lattr,
                           const float* __restrict__ alS, const float* __restrict__ alD,
                           const float* __restrict__ weV, unsigned* m, int E, int n) {
    int idx = blockIdx.x * blockDim.x + threadIdx.x;
    if (idx >= (E + n) * H) return;
    int e = idx / H, h = idx % H, s, d;
    float al = gat_alpha<H>(e, h, E, src, dst, ea, lattr, alS, alD, weV, s, d);
    atomicMax(&m[d * H + h], fenc(al));
}

template <int H>
__global__ void k_edge_den(const int* __restrict__ src, const int* __restrict__ dst,
                           const float* __restrict__ ea, const float* __restrict__ lattr,
                           const float* __restrict__ alS, const float* __restrict__ alD,
                           const float* __restrict__ weV, const unsigned* __restrict__ m,
                           float* den, int E, int n) {
    int idx = blockIdx.x * blockDim.x + threadIdx.x;
    if (idx >= (E + n) * H) return;
    int e = idx / H, h = idx % H, s, d;
    float al = gat_alpha<H>(e, h, E, src, dst, ea, lattr, alS, alD, weV, s, d);
    atomicAdd(&den[d * H + h], expf(al - fdec(m[d * H + h])));
}

template <int H, int C>
__global__ void k_edge_agg(const int* __restrict__ src, const int* __restrict__ dst,
                           const float* __restrict__ ea, const float* __restrict__ lattr,
                           const float* __restrict__ alS, const float* __restrict__ alD,
                           const float* __restrict__ weV, const unsigned* __restrict__ m,
                           const float* __restrict__ den, const float* __restrict__ T,
                           float* OUT, int E, int n) {
    int idx = blockIdx.x * blockDim.x + threadIdx.x;
    if (idx >= (E + n) * H) return;
    int e = idx / H, h = idx % H, s, d;
    float al = gat_alpha<H>(e, h, E, src, dst, ea, lattr, alS, alD, weV, s, d);
    float att = expf(al - fdec(m[d * H + h])) / den[d * H + h];
    const float* ts = T + s * H * C + h * C;
    float* od = OUT + d * H * C + h * C;
#pragma unroll
    for (int c = 0; c < C; ++c) atomicAdd(&od[c], att * ts[c]);
}

// ---------- bias + activations ----------
__global__ void k_bias_elu(float* Y, const float* b, int n, int F) {
    int i = blockIdx.x * blockDim.x + threadIdx.x;
    if (i >= n * F) return;
    float v = Y[i] + b[i % F];
    Y[i] = v > 0.f ? v : expm1f(v);
}
__global__ void k_bias_sig(float* Y, const float* b, int n) {
    int i = blockIdx.x * blockDim.x + threadIdx.x;
    if (i >= n) return;
    Y[i] = 1.f / (1.f + expf(-(Y[i] + b[0])));
}

// ---------- f32 -> f16 ----------
__global__ void k_cvt_f16(const float* X, _Float16* Y, int n) {
    int i = blockIdx.x * blockDim.x + threadIdx.x;
    if (i < n) Y[i] = (_Float16)X[i];
}

// ---------- layer-2 GEMM: [M x 128] f16 @ [128 x 64] f16 -> f32, via WMMA ----------
// LDS strides padded to 136 halves (272 B): half-wave lanes step 4 banks each,
// a b128 spans 4 banks -> the 16 lanes tile all 64 banks conflict-free.
#define SA_STRIDE 136
#define SB_STRIDE 136

__global__ __launch_bounds__(256) void k_gemm_wmma(const _Float16* __restrict__ A,
                                                   const _Float16* __restrict__ B,
                                                   float* __restrict__ C, int M) {
    __shared__ _Float16 sA [32 * SA_STRIDE];   // 8.7 KB, row-major [row][k]
    __shared__ _Float16 sBt[64 * SB_STRIDE];   // 17.4 KB, transposed [n][k]
    const int tid = threadIdx.x;
    const int rowBase = blockIdx.x * 32;

    // prefetch next block's A tile (global_prefetch_b8)
    if (rowBase + 32 < M) __builtin_prefetch(A + (rowBase + 32) * 128, 0, 1);

    for (int i = tid; i < 128 * 64; i += 256) {
        int k = i >> 6, n = i & 63;
        sBt[n * SB_STRIDE + k] = B[i];
    }
    for (int i = tid; i < 32 * 128; i += 256) {
        int r = i >> 7, k = i & 127;
        int row = rowBase + r;
        sA[r * SA_STRIDE + k] = (row < M) ? A[row * 128 + k] : (_Float16)0.f;
    }
    __syncthreads();

    const int wave = tid >> 5, lane = tid & 31;
    const int rowTile = (wave >> 2) * 16;     // 0 or 16
    const int colTile = (wave & 3) * 16;      // 0/16/32/48
    const int g  = lane >> 4;                 // lane half-group
    const int lm = lane & 15;

    const _Float16* aBase = &sA [(rowTile + lm) * SA_STRIDE + g * 16];
    const _Float16* bBase = &sBt[(colTile + lm) * SB_STRIDE + g * 16];

    v8f acc = {};
#pragma unroll
    for (int kt = 0; kt < 4; ++kt) {
        const int kb = kt * 32;
        // A 16x32 f16 k order used here (B matches): lane halves i -> K = g*16 + i (contiguous 16)
        v8h a0 = *(const v8h*)(aBase + kb);
        v8h a1 = *(const v8h*)(aBase + kb + 8);
        v16h a = __builtin_shufflevector(a0, a1, 0, 1, 2, 3, 4, 5, 6, 7,
                                                 8, 9, 10, 11, 12, 13, 14, 15);
        // B 32x16 f16 k order used here (same as A): lane halves i -> K = g*16 + i (contiguous 16)
        v8h b0 = *(const v8h*)(bBase + kb);
        v8h b1 = *(const v8h*)(bBase + kb + 8);
        v16h b = __builtin_shufflevector(b0, b1, 0, 1, 2, 3, 4, 5, 6, 7,
                                                 8, 9, 10, 11, 12, 13, 14, 15);
        acc = __builtin_amdgcn_wmma_f32_16x16x32_f16(
                  false, a, false, b, (short)0, acc, false, false);
    }
    // C 16x16 f32 layout: VGPR r -> M = r + 8*g, N = lane&15
#pragma unroll
    for (int r = 0; r < 8; ++r) {
        int row = rowBase + rowTile + r + g * 8;
        if (row < M) C[row * 64 + colTile + lm] = acc[r];
    }
}

static inline int gridFor(long n) { return (int)((n + TPB - 1) / TPB); }

extern "C" void kernel_launch(void* const* d_in, const int* in_sizes, int n_in,
                              void* d_out, int out_size, void* d_ws, size_t ws_size,
                              hipStream_t stream) {
    (void)n_in; (void)out_size; (void)ws_size;
    const float* x   = (const float*)d_in[0];
    const int*   ei  = (const int*)d_in[1];
    const float* ea  = (const float*)d_in[2];
    const float* W1  = (const float*)d_in[3];
    const float* aS1 = (const float*)d_in[4];
    const float* aD1 = (const float*)d_in[5];
    const float* We1 = (const float*)d_in[6];
    const float* aE1 = (const float*)d_in[7];
    const float* b1  = (const float*)d_in[8];
    const float* W2  = (const float*)d_in[9];
    const float* aS2 = (const float*)d_in[10];
    const float* aD2 = (const float*)d_in[11];
    const float* We2 = (const float*)d_in[12];
    const float* aE2 = (const float*)d_in[13];
    const float* b2  = (const float*)d_in[14];
    const float* W3  = (const float*)d_in[15];
    const float* aS3 = (const float*)d_in[16];
    const float* aD3 = (const float*)d_in[17];
    const float* We3 = (const float*)d_in[18];
    const float* aE3 = (const float*)d_in[19];
    const float* b3  = (const float*)d_in[20];

    const int N = in_sizes[0] / 5;
    const int E = in_sizes[1] / 2;
    const int* src = ei;
    const int* dst = ei + E;

    // ---- workspace layout (bump allocator with region reuse) ----
    char* w = (char*)d_ws;
    float*    deg   = (float*)w;      w += (size_t)N * 4;
    float*    lattr = (float*)w;      w += (size_t)N * 8;
    float*    alS   = (float*)w;      w += (size_t)N * 16;   // up to 4 heads
    float*    alD   = (float*)w;      w += (size_t)N * 16;
    unsigned* mbuf  = (unsigned*)w;   w += (size_t)N * 16;
    float*    den   = (float*)w;      w += (size_t)N * 16;
    float*    weV   = (float*)w;      w += 256;
    _Float16* w2h   = (_Float16*)w;   w += 128 * 64 * 2;
    char* regA = w;                   w += (size_t)N * 512;  // t1 | {h1h, t2}
    char* regB = w;                                         // out1 | {out2, t3}
    float*    t1   = (float*)regA;
    _Float16* h1h  = (_Float16*)regA;
    float*    t2   = (float*)(regA + (size_t)N * 256);
    float*    out1 = (float*)regB;
    float*    out2 = (float*)regB;
    float*    t3   = (float*)(regB + (size_t)N * 256);
    float*    outf = (float*)d_out;

    float ninf = -3.0e38f; unsigned nu; memcpy(&nu, &ninf, 4);
    const unsigned ENC_NEG = ~nu;   // fenc(-3e38): sign bit set -> ~bits

    // ---- stage 0: self-loop edge_attr = mean incoming edge_attr ----
    k_fill_f32<<<gridFor(N), TPB, 0, stream>>>(deg, 0.f, N);
    k_fill_f32<<<gridFor(2 * N), TPB, 0, stream>>>(lattr, 0.f, 2 * N);
    k_deg_attr<<<gridFor(E), TPB, 0, stream>>>(dst, ea, deg, lattr, E);
    k_loop_div<<<gridFor(N), TPB, 0, stream>>>(lattr, deg, N);

    // ================= Layer 1: IN=5, H=4, C=32 =================
    {
        const int H = 4, C = 32, F = H * C;
        k_xform<<<gridFor((long)N * F), TPB, 0, stream>>>(x, W1, t1, N, 5, F);
        k_alcoef<4, 32><<<gridFor((long)N * H), TPB, 0, stream>>>(t1, aS1, aD1, alS, alD, N);
        k_wevec<<<1, 64, 0, stream>>>(We1, aE1, weV, H, C);
        k_fill_u32<<<gridFor((long)N * H), TPB, 0, stream>>>(mbuf, ENC_NEG, N * H);
        k_fill_f32<<<gridFor((long)N * H), TPB, 0, stream>>>(den, 0.f, N * H);
        k_fill_f32<<<gridFor((long)N * F), TPB, 0, stream>>>(out1, 0.f, N * F);
        int tot = (E + N) * H;
        k_edge_max<4><<<gridFor(tot), TPB, 0, stream>>>(src, dst, ea, lattr, alS, alD, weV, mbuf, E, N);
        k_edge_den<4><<<gridFor(tot), TPB, 0, stream>>>(src, dst, ea, lattr, alS, alD, weV, mbuf, den, E, N);
        k_edge_agg<4, 32><<<gridFor(tot), TPB, 0, stream>>>(src, dst, ea, lattr, alS, alD, weV, mbuf, den, t1, out1, E, N);
        k_bias_elu<<<gridFor((long)N * F), TPB, 0, stream>>>(out1, b1, N, F);
    }

    // ================= Layer 2: IN=128, H=2, C=32 (WMMA GEMM) =================
    {
        const int H = 2, C = 32, F = H * C;
        k_cvt_f16<<<gridFor((long)N * 128), TPB, 0, stream>>>(out1, h1h, N * 128);
        k_cvt_f16<<<gridFor(128 * 64), TPB, 0, stream>>>(W2, w2h, 128 * 64);
        k_gemm_wmma<<<(N + 31) / 32, 256, 0, stream>>>(h1h, w2h, t2, N);
        k_alcoef<2, 32><<<gridFor((long)N * H), TPB, 0, stream>>>(t2, aS2, aD2, alS, alD, N);
        k_wevec<<<1, 64, 0, stream>>>(We2, aE2, weV, H, C);
        k_fill_u32<<<gridFor((long)N * H), TPB, 0, stream>>>(mbuf, ENC_NEG, N * H);
        k_fill_f32<<<gridFor((long)N * H), TPB, 0, stream>>>(den, 0.f, N * H);
        k_fill_f32<<<gridFor((long)N * F), TPB, 0, stream>>>(out2, 0.f, N * F);
        int tot = (E + N) * H;
        k_edge_max<2><<<gridFor(tot), TPB, 0, stream>>>(src, dst, ea, lattr, alS, alD, weV, mbuf, E, N);
        k_edge_den<2><<<gridFor(tot), TPB, 0, stream>>>(src, dst, ea, lattr, alS, alD, weV, mbuf, den, E, N);
        k_edge_agg<2, 32><<<gridFor(tot), TPB, 0, stream>>>(src, dst, ea, lattr, alS, alD, weV, mbuf, den, t2, out2, E, N);
        k_bias_elu<<<gridFor((long)N * F), TPB, 0, stream>>>(out2, b2, N, F);
    }

    // ================= Layer 3: IN=64, H=1, C=1, mean+sigmoid =================
    {
        k_xform<<<gridFor((long)N), TPB, 0, stream>>>(out2, W3, t3, N, 64, 1);
        k_alcoef<1, 1><<<gridFor((long)N), TPB, 0, stream>>>(t3, aS3, aD3, alS, alD, N);
        k_wevec<<<1, 64, 0, stream>>>(We3, aE3, weV, 1, 1);
        k_fill_u32<<<gridFor((long)N), TPB, 0, stream>>>(mbuf, ENC_NEG, N);
        k_fill_f32<<<gridFor((long)N), TPB, 0, stream>>>(den, 0.f, N);
        k_fill_f32<<<gridFor((long)N), TPB, 0, stream>>>(outf, 0.f, N);
        int tot = E + N;
        k_edge_max<1><<<gridFor(tot), TPB, 0, stream>>>(src, dst, ea, lattr, alS, alD, weV, mbuf, E, N);
        k_edge_den<1><<<gridFor(tot), TPB, 0, stream>>>(src, dst, ea, lattr, alS, alD, weV, mbuf, den, E, N);
        k_edge_agg<1, 1><<<gridFor(tot), TPB, 0, stream>>>(src, dst, ea, lattr, alS, alD, weV, mbuf, den, t3, outf, E, N);
        k_bias_sig<<<gridFor((long)N), TPB, 0, stream>>>(outf, b3, N);
    }
}
